// DocREDModel_38895223832754
// MI455X (gfx1250) — compile-verified, matrix operand order checked by reading, not run on hardware
//
#include <hip/hip_runtime.h>

typedef __attribute__((ext_vector_type(16))) _Float16 v16h;
typedef __attribute__((ext_vector_type(8)))  _Float16 v8h;
typedef __attribute__((ext_vector_type(8)))  float    v8f;
typedef unsigned int u32;
typedef __attribute__((ext_vector_type(4))) u32 u32x4;
typedef __attribute__((ext_vector_type(8))) int i32x8;
typedef __attribute__((ext_vector_type(4))) int i32x4;

#define HDIM   768
#define BLK    64
#define BATCH  16
#define SEQ    2048
#define NENT   32
#define NMENT  4
#define NPAIR  128
#define NROWS  2048          // BATCH*NPAIR
#define LCLS   97
#define LPAD   112           // pad to 7 WMMA N-tiles
#define KT1    24            // HDIM/32  (K tiles, GEMM1)
#define NT1    48            // HDIM/16  (N tiles, GEMM1)
#define KT3    1536          // HDIM*BLK/32 (K tiles, classifier)
#define NT3    7             // LPAD/16
#define KSPLIT 4             // classifier K-split for occupancy
#define CCHUNK (KT3 / KSPLIT)        // 384 K-tiles per split
#define ZCOLS  192                   // 3 bilinear blocks * 64 cols per split

// --- WMMA fragment K pattern (ISA 7.12.2, 16-bit A/B, wave32) ---
__device__ __forceinline__ int kpat(int hi, int h) {
    int lo   = h & 7;
    int quad = ((h >> 3) << 1) | hi;
    return quad * 8 + lo;
}

// ---- Tensor Data Mover: issue a 2D tile load (rows x rowq, 8B units) ----
// D# layout per CDNA5 ISA. One issue per calling wave.
__device__ __forceinline__ void tdm_load_2d(u32 lds_off, const void* gaddr,
                                            u32 rowq,       // row length, 8B units
                                            u32 rows,       // #rows
                                            u32 strideq) {  // row stride, 8B units
    unsigned long long ga = (unsigned long long)gaddr;
    u32x4 g0;
    g0[0] = 1u;                                   // count=1, user mode
    g0[1] = lds_off;                              // lds_addr
    g0[2] = (u32)(ga & 0xFFFFFFFFull);            // global_addr[31:0]
    g0[3] = (u32)((ga >> 32) & 0x01FFFFFFull) | (2u << 30);  // ga[56:32] | type=2
    i32x8 g1;
    g1[0] = (int)(3u << 16);                      // mask=0, data_size=3 (8B)
    g1[1] = (int)((rowq & 0xFFFFu) << 16);        // tensor_dim0[15:0] -> [31:16]
    g1[2] = (int)(((rowq >> 16) & 0xFFFFu) | ((rows & 0xFFFFu) << 16)); // dim0 hi | dim1 lo
    g1[3] = (int)((rowq & 0xFFFFu) << 16);        // tile_dim0 -> [31:16] (dim1 hi = 0)
    g1[4] = (int)(rows & 0xFFFFu);                // tile_dim1 (tile_dim2 = 0)
    g1[5] = (int)strideq;                         // tensor_dim0_stride[31:0]
    g1[6] = 0;                                    // stride hi | dim1_stride lo
    g1[7] = 0;
    i32x4 gz4 = {0, 0, 0, 0};
    i32x8 gz8 = {0, 0, 0, 0, 0, 0, 0, 0};
    __builtin_amdgcn_tensor_load_to_lds(g0, g1, gz4, gz4, gz8, 0);
}

// Pack a row-major (KT*32 x ncols) f32 B matrix into per-fragment lane order,
// zero-padding columns >= ncols. dst halfs: ((kt*NT + nt)*32 + lane)*16 + h
__global__ __launch_bounds__(256)
void pack_b_kernel(const float* __restrict__ src, _Float16* __restrict__ dst,
                   int KT, int NT, int ncols) {
    int idx = blockIdx.x * blockDim.x + threadIdx.x;
    int total = KT * NT * 512;
    if (idx >= total) return;
    int h    = idx & 15;
    int lane = (idx >> 4) & 31;
    int tile = idx >> 9;
    int nt   = tile % NT;
    int kt   = tile / NT;
    int n = nt * 16 + (lane & 15);
    int k = kt * 32 + kpat(lane >> 4, h);
    float v = (n < ncols) ? src[(size_t)k * ncols + n] : 0.0f;
    dst[idx] = (_Float16)v;
}

// ent[b,e,:] = sum_m w_m * tf[b, pos[b,e,m], :], w_m = 0 iff pos repeats earlier
__global__ __launch_bounds__(256)
void entity_kernel(const float* __restrict__ tf, const int* __restrict__ pos,
                   float* __restrict__ ent) {
    int be = blockIdx.x;             // b*NENT + e
    int b  = be >> 5;
    const int* p = pos + be * NMENT;
    int p0 = p[0], p1 = p[1], p2 = p[2], p3 = p[3];
    float w1 = (p1 != p0) ? 1.f : 0.f;
    float w2 = (p2 != p0 && p2 != p1) ? 1.f : 0.f;
    float w3 = (p3 != p0 && p3 != p1 && p3 != p2) ? 1.f : 0.f;
    const float* base = tf + (size_t)b * SEQ * HDIM;
    const float* r0 = base + (size_t)p0 * HDIM;
    const float* r1 = base + (size_t)p1 * HDIM;
    const float* r2 = base + (size_t)p2 * HDIM;
    const float* r3 = base + (size_t)p3 * HDIM;
    float* e = ent + (size_t)be * HDIM;
    for (int h = threadIdx.x; h < HDIM; h += blockDim.x)
        e[h] = r0[h] + w1 * r1[h] + w2 * r2[h] + w3 * r3[h];
}

// Gather subj/obj rows and store directly in WMMA A-fragment order (f16).
__global__ __launch_bounds__(256)
void pair_pack_kernel(const float* __restrict__ ent, const int* __restrict__ ht,
                      _Float16* __restrict__ subjA, _Float16* __restrict__ objA) {
    int n  = blockIdx.x;             // 0..NROWS-1
    int b  = n >> 7;                 // / NPAIR
    int pp = n & 127;
    int e0 = ht[((size_t)b * NPAIR + pp) * 2 + 0];
    int e1 = ht[((size_t)b * NPAIR + pp) * 2 + 1];
    const float* s = ent + ((size_t)b * NENT + e0) * HDIM;
    const float* o = ent + ((size_t)b * NENT + e1) * HDIM;
    int mt = n >> 4, lm = n & 15;
    for (int k = threadIdx.x; k < HDIM; k += blockDim.x) {
        int kt = k >> 5, kk = k & 31;
        int quad = kk >> 3, lo = kk & 7;
        int lane = lm + ((quad & 1) << 4);
        int h    = ((quad >> 1) << 3) + lo;
        size_t di = ((size_t)(mt * KT1 + kt) * 32 + lane) * 16 + h;
        subjA[di] = (_Float16)s[k];
        objA[di]  = (_Float16)o[k];
    }
}

// Z = tanh(A @ B + bias), A pre-packed fragments (NROWSxH), B pre-packed (HxH).
// Block: 8 waves; each wave owns one 16x16 output tile. Grid: (NROWS/16, H/128).
__global__ __launch_bounds__(256)
void gemm_tanh_kernel(const _Float16* __restrict__ Ap, const _Float16* __restrict__ Bp,
                      const float* __restrict__ bias, _Float16* __restrict__ Z) {
    int lane = threadIdx.x & 31;
    int w    = threadIdx.x >> 5;
    int mt   = blockIdx.x;
    int nt   = blockIdx.y * 8 + w;
    v8f acc = {};
    const v16h* a_base = (const v16h*)Ap + (size_t)mt * KT1 * 32 + lane;
    const v16h* b_base = (const v16h*)Bp + (size_t)nt * 32 + lane;
    for (int kt = 0; kt < KT1; ++kt) {
        __builtin_prefetch((const void*)(b_base + (size_t)(kt + 2) * NT1 * 32), 0, 0);
        v16h a = a_base[(size_t)kt * 32];
        v16h b = b_base[(size_t)kt * NT1 * 32];
        acc = __builtin_amdgcn_wmma_f32_16x16x32_f16(false, a, false, b,
                                                     (short)0, acc, false, false);
    }
    int ncol = nt * 16 + (lane & 15);
    int hi   = lane >> 4;
    float bi = bias[ncol];
    for (int r = 0; r < 8; ++r) {
        int m = mt * 16 + r + hi * 8;
        Z[(size_t)m * HDIM + ncol] = (_Float16)tanhf(acc[r] + bi);
    }
}

// partial[ks] = (blockwise outer(z_s, z_o)) @ Wc over K-tiles [ks*384, ks*384+384).
// Block: 7 waves (one per padded N tile). Grid: (NROWS/16, KSPLIT).
// z_s/z_o slices (16 x 192 f16, strided) staged into LDS via the Tensor Data Mover.
__global__ __launch_bounds__(224)
void classifier_kernel(const _Float16* __restrict__ zs, const _Float16* __restrict__ zo,
                       const _Float16* __restrict__ WcP, float* __restrict__ partial) {
    __shared__ _Float16 zs_t[16 * ZCOLS];
    __shared__ _Float16 zo_t[16 * ZCOLS];
    int mt = blockIdx.x;
    int ks = blockIdx.y;

    if (threadIdx.x == 0) {
        // 16 rows x 384B tiles, global row stride 1536B, 8-byte units
        size_t goff = ((size_t)mt * 16 * HDIM + (size_t)ks * ZCOLS);
        tdm_load_2d((u32)(unsigned long long)(uintptr_t)zs_t, (const void*)(zs + goff),
                    ZCOLS * 2 / 8, 16, HDIM * 2 / 8);
        tdm_load_2d((u32)(unsigned long long)(uintptr_t)zo_t, (const void*)(zo + goff),
                    ZCOLS * 2 / 8, 16, HDIM * 2 / 8);
        __builtin_amdgcn_s_wait_tensorcnt(0);
    }
    __syncthreads();

    int lane = threadIdx.x & 31;
    int w    = threadIdx.x >> 5;         // N tile 0..6
    int lm   = lane & 15;                // A-fragment M index
    int hi   = lane >> 4;
    v8f acc = {};
    const v16h* bbase = (const v16h*)WcP + (size_t)w * 32 + lane;
    const _Float16* zsr = zs_t + (size_t)lm * ZCOLS;
    const v8h*      zor = (const v8h*)(zo_t + (size_t)lm * ZCOLS);

    int c0 = ks * CCHUNK;
    for (int cl = 0; cl < CCHUNK; ++cl) {
        int c = c0 + cl;
        __builtin_prefetch((const void*)(bbase + (size_t)(c + 4) * NT3 * 32), 0, 0);
        // flat K index c*32+kpat = kb*4096 + i*64 + j -> a = zs[m,kb*64+i]*zo[m,kb*64+j]
        int kbl = cl >> 7;                // local bilinear block 0..2
        int rem = cl & 127;
        int i   = rem >> 1;
        int j0  = (rem & 1) << 5;
        _Float16 s = zsr[kbl * 64 + i];
        int vb = (kbl * 64 + j0 + hi * 8) >> 3;   // v8h units, 16B aligned
        v8h lo8 = zor[vb];
        v8h hi8 = zor[vb + 2];
        v16h a;
        #pragma unroll
        for (int t = 0; t < 8; ++t) { a[t] = s * lo8[t]; a[8 + t] = s * hi8[t]; }
        v16h b = bbase[(size_t)c * NT3 * 32];
        acc = __builtin_amdgcn_wmma_f32_16x16x32_f16(false, a, false, b,
                                                     (short)0, acc, false, false);
    }
    int col = w * 16 + lm;
    float* prow = partial + (size_t)ks * NROWS * LPAD;
    for (int r = 0; r < 8; ++r) {
        int n = mt * 16 + r + hi * 8;
        prow[(size_t)n * LPAD + col] = acc[r];
    }
}

// out[n, col<97] = bc[col] + sum_ks partial[ks][n][col]   (deterministic reduce)
__global__ __launch_bounds__(256)
void reduce_kernel(const float* __restrict__ partial, const float* __restrict__ bc,
                   float* __restrict__ out) {
    int idx = blockIdx.x * blockDim.x + threadIdx.x;
    if (idx >= NROWS * LCLS) return;
    int n   = idx / LCLS;
    int col = idx - n * LCLS;
    float v = bc[col];
    for (int ks = 0; ks < KSPLIT; ++ks)
        v += partial[((size_t)ks * NROWS + n) * LPAD + col];
    out[idx] = v;
}

extern "C" void kernel_launch(void* const* d_in, const int* in_sizes, int n_in,
                              void* d_out, int out_size, void* d_ws, size_t ws_size,
                              hipStream_t stream) {
    (void)in_sizes; (void)n_in; (void)out_size; (void)ws_size;
    const float* tf  = (const float*)d_in[0];
    const int*   pos = (const int*)  d_in[1];
    const int*   ht  = (const int*)  d_in[2];
    const float* Wh  = (const float*)d_in[3];
    const float* bh  = (const float*)d_in[4];
    const float* Wt  = (const float*)d_in[5];
    const float* bt  = (const float*)d_in[6];
    const float* Wc  = (const float*)d_in[7];
    const float* bc  = (const float*)d_in[8];
    float* out = (float*)d_out;

    char* ws = (char*)d_ws;
    size_t off = 0;
    float*    ent   = (float*)(ws + off);    off += (size_t)BATCH * NENT * HDIM * sizeof(float);
    _Float16* subjA = (_Float16*)(ws + off); off += (size_t)NROWS * HDIM * sizeof(_Float16);
    _Float16* objA  = (_Float16*)(ws + off); off += (size_t)NROWS * HDIM * sizeof(_Float16);
    _Float16* WhP   = (_Float16*)(ws + off); off += (size_t)KT1 * NT1 * 512 * sizeof(_Float16);
    _Float16* WtP   = (_Float16*)(ws + off); off += (size_t)KT1 * NT1 * 512 * sizeof(_Float16);
    _Float16* WcP   = (_Float16*)(ws + off); off += (size_t)KT3 * NT3 * 512 * sizeof(_Float16);
    _Float16* zsb   = (_Float16*)(ws + off); off += (size_t)NROWS * HDIM * sizeof(_Float16);
    _Float16* zob   = (_Float16*)(ws + off); off += (size_t)NROWS * HDIM * sizeof(_Float16);
    float*    part  = (float*)(ws + off);    off += (size_t)KSPLIT * NROWS * LPAD * sizeof(float);

    // 1) pack B matrices into WMMA fragment order (Wc zero-padded 97->112)
    int tot1 = KT1 * NT1 * 512;
    int tot3 = KT3 * NT3 * 512;
    pack_b_kernel<<<(tot1 + 255) / 256, 256, 0, stream>>>(Wh, WhP, KT1, NT1, HDIM);
    pack_b_kernel<<<(tot1 + 255) / 256, 256, 0, stream>>>(Wt, WtP, KT1, NT1, HDIM);
    pack_b_kernel<<<(tot3 + 255) / 256, 256, 0, stream>>>(Wc, WcP, KT3, NT3, LCLS);

    // 2) dedup-weighted entity aggregation
    entity_kernel<<<BATCH * NENT, 256, 0, stream>>>(tf, pos, ent);

    // 3) pair gather, packed as WMMA A fragments
    pair_pack_kernel<<<NROWS, 256, 0, stream>>>(ent, ht, subjA, objA);

    // 4) z_s / z_o GEMMs with fused bias + tanh (f16 WMMA, f32 accumulate)
    dim3 g1(NROWS / 16, HDIM / 128);
    gemm_tanh_kernel<<<g1, 256, 0, stream>>>(subjA, WhP, bh, zsb);
    gemm_tanh_kernel<<<g1, 256, 0, stream>>>(objA,  WtP, bt, zob);

    // 5) bilinear classifier, K-split x4 for occupancy, TDM-staged LDS tiles
    dim3 g3(NROWS / 16, KSPLIT);
    classifier_kernel<<<g3, 224, 0, stream>>>(zsb, zob, WcP, part);

    // 6) deterministic partial reduce + bias
    reduce_kernel<<<(NROWS * LCLS + 255) / 256, 256, 0, stream>>>(part, bc, out);
}
